// OldSelfAttention_2843268350495
// MI455X (gfx1250) — matrix operand order repaired and checked by reading, not compiled, not run on hardware
//
#include <hip/hip_runtime.h>

// ---------------------------------------------------------------------------
// Self-attention forward for MI455X (gfx1250, wave32, WMMA).
//   HIDDEN=1024, HEADS=16, HEAD_DIM=64, B=2, S=2048.
// Pipeline:
//   cvt_x     : f32 hidden -> f16
//   cvt_wt    : f32 W[k][n] -> f16 Wt[n][k] (K-contiguous for B fragments)
//   qkv_gemm  : X(f16) @ Wt -> Q[s][d], K[s][d], Vt[d][s]; 32x64 wave tile,
//               ping-pong double-buffered (no register rotation -> no
//               v_mov/v_nop hazard traffic between WMMAs).
//   attn      : flash attention, 32 queries/wave, v_wmma_f32_16x16x32_f16
//               for QK^T and P*V, online softmax in registers, out = f32.
// Workspace: 8 + 6 + 8 + 8 + 8 = 38 MB.
// ---------------------------------------------------------------------------

#define HIDDEN   1024
#define HEADS    16
#define HEAD_DIM 64
#define BATCH    2
#define SEQ      2048
#define MTOT     (BATCH * SEQ)   // 4096 rows

typedef __attribute__((ext_vector_type(16))) _Float16 v16h;
typedef __attribute__((ext_vector_type(8)))  _Float16 v8h;
typedef __attribute__((ext_vector_type(8)))  float    v8f;

// Build a 16-half WMMA fragment from two contiguous 16-byte chunks.
__device__ __forceinline__ v16h load_frag(const _Float16* p0, const _Float16* p1) {
  v16h r;
  *reinterpret_cast<v8h*>(&r)       = *reinterpret_cast<const v8h*>(p0);
  *(reinterpret_cast<v8h*>(&r) + 1) = *reinterpret_cast<const v8h*>(p1);
  return r;
}

// ---------------------------------------------------------------------------
__global__ void cvt_x(const float* __restrict__ x, _Float16* __restrict__ y, int n) {
  int i = blockIdx.x * blockDim.x + threadIdx.x;
  if (i < n) y[i] = (_Float16)x[i];
}

// Wt[mat][n][k] = (f16) W[mat][k][n]   (blockIdx.y selects Wq/Wk/Wv)
__global__ void cvt_wt(const float* __restrict__ Wq, const float* __restrict__ Wk,
                       const float* __restrict__ Wv, _Float16* __restrict__ Wt) {
  const float* W = (blockIdx.y == 0) ? Wq : ((blockIdx.y == 1) ? Wk : Wv);
  _Float16* dst = Wt + (size_t)blockIdx.y * HIDDEN * HIDDEN;
  int i = blockIdx.x * blockDim.x + threadIdx.x;   // exactly HIDDEN*HIDDEN threads
  int n = i >> 10, k = i & 1023;
  dst[i] = (_Float16)W[k * HIDDEN + n];
}

// ---------------------------------------------------------------------------
// QKV GEMM: each wave owns a 32x64 tile (2 A row-frags x 4 B col-frags,
// 8 f32 accumulators). Ping-pong software pipeline: phase p computes from
// buffer p while the other buffer is being filled for the next chunk.
__global__ void __launch_bounds__(256)
qkv_gemm(const _Float16* __restrict__ X, const _Float16* __restrict__ Wt,
         const float* __restrict__ bq, const float* __restrict__ bk,
         const float* __restrict__ bv,
         _Float16* __restrict__ Q, _Float16* __restrict__ K,
         _Float16* __restrict__ Vt) {
  const int lane = threadIdx.x & 31;
  const int wid  = threadIdx.x >> 5;
  const int gw   = blockIdx.x * 8 + wid;        // 6144 waves total
  const int mat  = gw / (128 * 16);             // 0=Q 1=K 2=V
  const int rem  = gw % (128 * 16);
  const int m0   = (rem >> 4) * 32;             // row tile base (0..4064)
  const int n0   = (rem & 15) * 64;             // col group base (0..960)
  const int hi   = lane >> 4;
  const int lc   = lane & 15;

  const _Float16* W    = Wt + (size_t)mat * HIDDEN * HIDDEN;
  const float*    bias = (mat == 0) ? bq : ((mat == 1) ? bk : bv);

  v8f acc[2][4] = {};
  const _Float16* xrow[2] = { X + (size_t)(m0 + lc) * HIDDEN,
                              X + (size_t)(m0 + 16 + lc) * HIDDEN };
  const _Float16* wbase = W + (size_t)(n0 + lc) * HIDDEN + hi * 8;

  v16h aB[2][2], bB[2][4];   // [phase][frag]

#define LOAD_PHASE(p, koff)                                                    \
  do {                                                                         \
    _Pragma("unroll")                                                          \
    for (int r = 0; r < 2; ++r)                                                \
      aB[p][r] = load_frag(xrow[r] + (koff) + hi * 8,                          \
                           xrow[r] + (koff) + hi * 8 + 16);                    \
    _Pragma("unroll")                                                          \
    for (int t = 0; t < 4; ++t) {                                              \
      const _Float16* wr = wbase + (size_t)t * 16 * HIDDEN + (koff);           \
      bB[p][t] = load_frag(wr, wr + 16);                                       \
    }                                                                          \
  } while (0)

#define WMMA_PHASE(p)                                                          \
  do {                                                                         \
    _Pragma("unroll")                                                          \
    for (int t = 0; t < 4; ++t) {                                              \
      acc[0][t] = __builtin_amdgcn_wmma_f32_16x16x32_f16(                      \
          false, aB[p][0], false, bB[p][t], (short)0, acc[0][t], false, false);\
      acc[1][t] = __builtin_amdgcn_wmma_f32_16x16x32_f16(                      \
          false, aB[p][1], false, bB[p][t], (short)0, acc[1][t], false, false);\
    }                                                                          \
  } while (0)

  LOAD_PHASE(0, 0);
  // Steady state: 15 iterations cover chunks 0..29 (two per iteration).
  for (int kk = 0; kk < HIDDEN - 64; kk += 64) {
    LOAD_PHASE(1, kk + 32);   // fill other buffer while phase-0 computes
    WMMA_PHASE(0);
    LOAD_PHASE(0, kk + 64);   // fill buffer 0 for chunk after next
    WMMA_PHASE(1);
  }
  // Tail: chunks 30 (kk=960, phase 0) and 31 (kk=992, phase 1).
  LOAD_PHASE(1, HIDDEN - 32);
  WMMA_PHASE(0);
  WMMA_PHASE(1);

#undef LOAD_PHASE
#undef WMMA_PHASE

  const int b    = m0 >> 11;        // batch (32-row tiles never cross batch)
  const int head = n0 >> 6;
  const int bh   = b * HEADS + head;

#pragma unroll
  for (int r = 0; r < 2; ++r) {
    const int s0 = (m0 & 2047) + r * 16;
#pragma unroll
    for (int t = 0; t < 4; ++t) {
      const int d  = t * 16 + lc;
      const float bb = bias[n0 + t * 16 + lc];
      if (mat == 2) {
        // Vt[bh][d][s] : 8 consecutive s per lane-half -> one b128 store
        v8h hv;
#pragma unroll
        for (int j = 0; j < 8; ++j) hv[j] = (_Float16)(acc[r][t][j] + bb);
        *reinterpret_cast<v8h*>(Vt + ((size_t)bh * HEAD_DIM + d) * SEQ + s0 + hi * 8) = hv;
      } else {
        _Float16* dst = (mat == 0) ? Q : K;
#pragma unroll
        for (int j = 0; j < 8; ++j) {
          const int s = s0 + hi * 8 + j;
          dst[((size_t)bh * SEQ + s) * HEAD_DIM + d] = (_Float16)(acc[r][t][j] + bb);
        }
      }
    }
  }
}

// ---------------------------------------------------------------------------
// Flash attention: one wave = one (batch,head) x 32-query tile (2 q-subtiles
// sharing every K/V fragment). Per 32-key chunk: 8 score WMMAs, online
// softmax, LDS transpose of P, 8 context WMMAs.
__global__ void __launch_bounds__(256)
attn(const _Float16* __restrict__ Q, const _Float16* __restrict__ K,
     const _Float16* __restrict__ Vt, const float* __restrict__ head_mask,
     float* __restrict__ out) {
  // Per-wave 2x(16x32) f16 P tiles; row stride 48 halfs (96B, b128-aligned).
  __shared__ _Float16 plds[8][2][16 * 48];   // 24 KB / block

  const int lane = threadIdx.x & 31;
  const int wid  = threadIdx.x >> 5;
  const int gw   = blockIdx.x * 8 + wid;   // 2048 waves total
  const int bh   = gw >> 6;                // (b*HEADS + head), 64 q-blocks each
  const int q0   = (gw & 63) * 32;
  const int hi   = lane >> 4;
  const int lc   = lane & 15;
  const int head = bh & (HEADS - 1);
  const int b    = bh >> 4;

  const _Float16* Qb = Q  + (size_t)bh * SEQ * HEAD_DIM;
  const _Float16* Kb = K  + (size_t)bh * SEQ * HEAD_DIM;
  const _Float16* Vb = Vt + (size_t)bh * HEAD_DIM * SEQ;

  // Loop-invariant Q fragments: [q-subtile][d-half]
  v16h aQ[2][2];
#pragma unroll
  for (int qt = 0; qt < 2; ++qt) {
    const _Float16* qrow = Qb + (size_t)(q0 + qt * 16 + lc) * HEAD_DIM;
#pragma unroll
    for (int w = 0; w < 2; ++w)
      aQ[qt][w] = load_frag(qrow + w * 32 + hi * 8, qrow + w * 32 + hi * 8 + 16);
  }

  v8f ctx[2][4] = {};
  float mrow[2][8], lrow[2][8];
#pragma unroll
  for (int qt = 0; qt < 2; ++qt)
#pragma unroll
    for (int j = 0; j < 8; ++j) { mrow[qt][j] = -__builtin_inff(); lrow[qt][j] = 0.f; }

  for (int k0 = 0; k0 < SEQ; k0 += 32) {
    // ---- K fragments for this chunk: [key-subtile][d-half] ----
    v16h bK[2][2];
#pragma unroll
    for (int t = 0; t < 2; ++t) {
      const _Float16* krow = Kb + (size_t)(k0 + t * 16 + lc) * HEAD_DIM;
#pragma unroll
      for (int w = 0; w < 2; ++w)
        bK[t][w] = load_frag(krow + w * 32 + hi * 8, krow + w * 32 + hi * 8 + 16);
    }
    // ---- scores: 8 WMMAs (2 q-subtiles x 2 key-subtiles x 2 d-halves) ----
    v8f sc[2][2];
#pragma unroll
    for (int qt = 0; qt < 2; ++qt)
#pragma unroll
      for (int t = 0; t < 2; ++t) {
        v8f c = {};
#pragma unroll
        for (int w = 0; w < 2; ++w)
          c = __builtin_amdgcn_wmma_f32_16x16x32_f16(false, aQ[qt][w], false, bK[t][w],
                                                     (short)0, c, false, false);
        sc[qt][t] = c;
      }

    // ---- V fragments: issued now so they overlap the softmax VALU work ----
    v16h bV[4];
#pragma unroll
    for (int t = 0; t < 4; ++t) {
      const _Float16* vrow = Vb + (size_t)(t * 16 + lc) * SEQ + k0 + hi * 8;
      bV[t] = load_frag(vrow, vrow + 16);
    }

    // ---- online softmax per q-subtile (row stats per lane-half) ----
#pragma unroll
    for (int qt = 0; qt < 2; ++qt) {
      _Float16* pl = &plds[wid][qt][0];
#pragma unroll
      for (int j = 0; j < 8; ++j) {
        float s0 = sc[qt][0][j] * 0.125f;      // 1/sqrt(64)
        float s1 = sc[qt][1][j] * 0.125f;
        float cm = fmaxf(s0, s1);
#pragma unroll
        for (int off = 1; off < 16; off <<= 1)
          cm = fmaxf(cm, __shfl_xor(cm, off, 32));   // stays within 16-lane half
        float mnew  = fmaxf(mrow[qt][j], cm);
        float alpha = __expf(mrow[qt][j] - mnew);
        float p0 = __expf(s0 - mnew);
        float p1 = __expf(s1 - mnew);
        float rs = p0 + p1;
#pragma unroll
        for (int off = 1; off < 16; off <<= 1)
          rs += __shfl_xor(rs, off, 32);
        lrow[qt][j] = lrow[qt][j] * alpha + rs;
        mrow[qt][j] = mnew;
#pragma unroll
        for (int t = 0; t < 4; ++t) ctx[qt][t][j] *= alpha;
        // stash P (C layout) into LDS tile: row = j + 8*hi, cols lc / lc+16
        const int r = j + hi * 8;
        pl[r * 48 + lc]      = (_Float16)p0;
        pl[r * 48 + 16 + lc] = (_Float16)p1;
      }
    }
    asm volatile("s_wait_dscnt 0x0" ::: "memory");  // cross-lane LDS handoff

    // Re-read P tiles as A fragments (16x32, Kdim = keys)
    v16h aP[2];
#pragma unroll
    for (int qt = 0; qt < 2; ++qt) {
      const _Float16* prow = &plds[wid][qt][0] + lc * 48;
      aP[qt] = load_frag(prow + hi * 8, prow + hi * 8 + 16);
    }

    // ---- ctx += P * V : 8 WMMAs ----
#pragma unroll
    for (int qt = 0; qt < 2; ++qt)
#pragma unroll
      for (int t = 0; t < 4; ++t)
        ctx[qt][t] = __builtin_amdgcn_wmma_f32_16x16x32_f16(false, aP[qt], false, bV[t],
                                                            (short)0, ctx[qt][t], false, false);
  }

  // ---- normalize, apply post-softmax head mask, write [b][s][h*64+d] ----
  const float hm = head_mask[head];
#pragma unroll
  for (int qt = 0; qt < 2; ++qt)
#pragma unroll
    for (int j = 0; j < 8; ++j) {
      const float inv = hm / lrow[qt][j];
      const int s = q0 + qt * 16 + hi * 8 + j;
      float* orow = out + ((size_t)(b * SEQ + s)) * HIDDEN + head * HEAD_DIM;
#pragma unroll
      for (int t = 0; t < 4; ++t)
        orow[t * 16 + lc] = ctx[qt][t][j] * inv;
    }
}

// ---------------------------------------------------------------------------
extern "C" void kernel_launch(void* const* d_in, const int* in_sizes, int n_in,
                              void* d_out, int out_size, void* d_ws, size_t ws_size,
                              hipStream_t stream) {
  const float* hidden    = (const float*)d_in[0];
  const float* head_mask = (const float*)d_in[1];
  const float* Wq = (const float*)d_in[2];
  const float* bq = (const float*)d_in[3];
  const float* Wk = (const float*)d_in[4];
  const float* bk = (const float*)d_in[5];
  const float* Wv = (const float*)d_in[6];
  const float* bv = (const float*)d_in[7];
  float* out = (float*)d_out;

  // Workspace carve-up (38 MB total)
  char* ws = (char*)d_ws;
  _Float16* X16  = (_Float16*)ws;  ws += (size_t)MTOT * HIDDEN * 2;       // 8 MB
  _Float16* Wt   = (_Float16*)ws;  ws += (size_t)3 * HIDDEN * HIDDEN * 2; // 6 MB
  _Float16* Q16  = (_Float16*)ws;  ws += (size_t)MTOT * HIDDEN * 2;       // 8 MB
  _Float16* K16  = (_Float16*)ws;  ws += (size_t)MTOT * HIDDEN * 2;       // 8 MB
  _Float16* Vt16 = (_Float16*)ws;                                         // 8 MB

  cvt_x<<<(MTOT * HIDDEN) / 256, 256, 0, stream>>>(hidden, X16, MTOT * HIDDEN);
  dim3 gt(HIDDEN * HIDDEN / 256, 3);
  cvt_wt<<<gt, 256, 0, stream>>>(Wq, Wk, Wv, Wt);
  // 3 mats * 128 M-tiles(32 rows) * 16 N-groups = 6144 waves / 8 per block
  qkv_gemm<<<(3 * 128 * 16) / 8, 256, 0, stream>>>(X16, Wt, bq, bk, bv,
                                                   Q16, K16, Vt16);
  // B*H*(S/32) = 2048 waves / 8 per block
  attn<<<(BATCH * HEADS * (SEQ / 32)) / 8, 256, 0, stream>>>(Q16, K16, Vt16,
                                                             head_mask, out);
}
